// AttentionLSTM_63436666962433
// MI455X (gfx1250) — matrix operand order repaired and checked by reading, not compiled, not run on hardware
//
#include <hip/hip_runtime.h>
#include <math.h>

// Problem constants (reference): N=128, T=64, D=1024, H=1024, P=16, 4H=4096
#define NB     128
#define TT     64
#define DD     1024
#define HH     1024
#define PP     16
#define FH     4096           // 4*H
#define KC     3072           // concat K = H(h) + H(attn) + D(xt)
#define KT32   96             // KC / 32
#define KSPLIT 4
#define KTP    (KT32 / KSPLIT)   // 24 kt iterations per split
#define KW     (KTP * 32)        // 768 K values per split

typedef __attribute__((ext_vector_type(16))) __bf16 bf16x16;
typedef __attribute__((ext_vector_type(8)))  float  v8f;
typedef __attribute__((ext_vector_type(4)))  unsigned int u32x4;

union FragU { bf16x16 v; u32x4 u[2]; };

__device__ __forceinline__ float sigmoidf(float x) { return 1.0f / (1.0f + expf(-x)); }

// CDNA5 async global->LDS copy, 16 bytes per lane (VGLOBAL encoding:
// VDST = LDS byte address VGPR, VADDR = 64-bit global address). ASYNCcnt.
__device__ __forceinline__ void async_copy_b128(unsigned lds_off, const void* gaddr) {
    asm volatile("global_load_async_to_lds_b128 %0, %1, off"
                 :: "v"(lds_off), "v"(gaddr) : "memory");
}
__device__ __forceinline__ void wait_async0() {
    asm volatile("s_wait_asynccnt 0x0" ::: "memory");
}

// ---------------------------------------------------------------------------
// Pack [Wh; Wattn; Wx] (each 1024x4096 f32 row-major) into bf16 WMMA B-fragment
// layout: fragment (kt, jt) of shape 32(K)x16(N); within fragment, element
// [lane*16 + i] = W[kt*32 + (lane>=16?16:0) + i, jt*16 + (lane&15)]
// (contiguous k order, same as A: lanes 0-15 hold K 0..15, lanes 16-31 hold K 16..31)
// ---------------------------------------------------------------------------
__global__ void pack_weights_kernel(const float* __restrict__ Wx,
                                    const float* __restrict__ Wh,
                                    const float* __restrict__ Wattn,
                                    __bf16* __restrict__ Wp) {
    long idx = (long)blockIdx.x * 256 + threadIdx.x;   // one per packed elem
    long frag   = idx >> 9;          // / 512
    int  within = (int)(idx & 511);
    int  lane   = within >> 4;
    int  i      = within & 15;
    int  kt     = (int)(frag % KT32);
    long jt     = frag / KT32;
    int  k      = kt * 32 + ((lane >= 16) ? 16 : 0) + i;
    int  j      = (int)(jt * 16) + (lane & 15);
    const float* src;
    int kr;
    if (k < HH)           { src = Wh;    kr = k; }
    else if (k < 2 * HH)  { src = Wattn; kr = k - HH; }
    else                  { src = Wx;    kr = k - 2 * HH; }
    Wp[idx] = (__bf16)src[(long)kr * FH + j];
}

// ---------------------------------------------------------------------------
// h0 = c0 = mean of A over the 4x4 grid (P=16 contiguous floats per (n,h))
// ---------------------------------------------------------------------------
__global__ void init_state_kernel(const float* __restrict__ Af,
                                  float* __restrict__ hbuf,
                                  float* __restrict__ cbuf) {
    long idx = (long)blockIdx.x * 256 + threadIdx.x;   // n*H + h
    const float* a = Af + idx * PP;
    float s = 0.f;
#pragma unroll
    for (int p = 0; p < PP; ++p) s += a[p];
    s *= (1.0f / 16.0f);
    hbuf[idx] = s;
    cbuf[idx] = s;
}

// ---------------------------------------------------------------------------
// Per-step attention + operand pack.  One block per sample n (256 thr, 8 waves)
// ---------------------------------------------------------------------------
__global__ void attn_pack_kernel(const float* __restrict__ Af,
                                 const float* __restrict__ x,
                                 const float* __restrict__ hbuf,
                                 __bf16* __restrict__ hy,
                                 int t) {
    int n   = blockIdx.x;
    int tid = threadIdx.x;
    __shared__ float red[8 * PP];
    __shared__ float sc[PP];
    __shared__ float w[PP];

    const float* An = Af + (long)n * HH * PP;
    const float* hn = hbuf + (long)n * HH;

    float loc[PP];
#pragma unroll
    for (int p = 0; p < PP; ++p) loc[p] = 0.f;

    for (int h = tid; h < HH; h += 256) {
        float hv = hn[h];
        const float4* ar = (const float4*)(An + (long)h * PP);
        float4 a0 = ar[0], a1 = ar[1], a2 = ar[2], a3 = ar[3];
        loc[0]  += hv * a0.x;  loc[1]  += hv * a0.y;  loc[2]  += hv * a0.z;  loc[3]  += hv * a0.w;
        loc[4]  += hv * a1.x;  loc[5]  += hv * a1.y;  loc[6]  += hv * a1.z;  loc[7]  += hv * a1.w;
        loc[8]  += hv * a2.x;  loc[9]  += hv * a2.y;  loc[10] += hv * a2.z;  loc[11] += hv * a2.w;
        loc[12] += hv * a3.x;  loc[13] += hv * a3.y;  loc[14] += hv * a3.z;  loc[15] += hv * a3.w;
    }
    // wave32 reduction
#pragma unroll
    for (int p = 0; p < PP; ++p)
#pragma unroll
        for (int off = 16; off > 0; off >>= 1)
            loc[p] += __shfl_down(loc[p], off, 32);
    if ((tid & 31) == 0) {
        int wv = tid >> 5;
#pragma unroll
        for (int p = 0; p < PP; ++p) red[wv * PP + p] = loc[p];
    }
    __syncthreads();
    if (tid < PP) {
        float s = 0.f;
#pragma unroll
        for (int wv = 0; wv < 8; ++wv) s += red[wv * PP + tid];
        sc[tid] = s * (1.0f / 32.0f);      // / sqrt(H)
    }
    __syncthreads();
    if (tid == 0) {
        float mx = -1e30f;
#pragma unroll
        for (int p = 0; p < PP; ++p) mx = fmaxf(mx, sc[p]);
        float sum = 0.f;
#pragma unroll
        for (int p = 0; p < PP; ++p) { float e = expf(sc[p] - mx); w[p] = e; sum += e; }
        float inv = 1.0f / sum;
#pragma unroll
        for (int p = 0; p < PP; ++p) w[p] *= inv;
    }
    __syncthreads();

    __bf16* hyn = hy + (long)n * KC;
    const float* xt = x + ((long)n * TT + t) * DD;
    for (int h = tid; h < HH; h += 256) {
        const float4* ar = (const float4*)(An + (long)h * PP);
        float4 a0 = ar[0], a1 = ar[1], a2 = ar[2], a3 = ar[3];
        float s = a0.x * w[0]  + a0.y * w[1]  + a0.z * w[2]  + a0.w * w[3]
                + a1.x * w[4]  + a1.y * w[5]  + a1.z * w[6]  + a1.w * w[7]
                + a2.x * w[8]  + a2.y * w[9]  + a2.z * w[10] + a2.w * w[11]
                + a3.x * w[12] + a3.y * w[13] + a3.z * w[14] + a3.w * w[15];
        hyn[h]          = (__bf16)hn[h];   // h
        hyn[HH + h]     = (__bf16)s;       // attn
        hyn[2 * HH + h] = (__bf16)xt[h];   // xt
    }
}

// ---------------------------------------------------------------------------
// Fragment helpers
// ---------------------------------------------------------------------------
__device__ __forceinline__ void load_b4(FragU b[4], const __bf16* __restrict__ Wp,
                                        long jt0, int kt, int lane) {
#pragma unroll
    for (int tj = 0; tj < 4; ++tj) {
        const __bf16* bp = Wp + (((jt0 + tj) * KT32 + kt) << 9) + lane * 16;
        b[tj].u[0] = *(const u32x4*)(bp);
        b[tj].u[1] = *(const u32x4*)(bp + 8);
    }
}

__device__ __forceinline__ void load_a_lds(FragU& a, const __bf16* alrow, int itk, int half) {
    int kk = itk * 32 + (half ? 16 : 0);
    a.u[0] = *(const u32x4*)(alrow + kk);          // -> ds_load_b128
    a.u[1] = *(const u32x4*)(alrow + kk + 8);
}

__device__ __forceinline__ void mm4(v8f acc[4], const FragU& a, const FragU b[4]) {
#pragma unroll
    for (int tj = 0; tj < 4; ++tj)
        acc[tj] = __builtin_amdgcn_wmma_f32_16x16x32_bf16(
            false, a.v, false, b[tj].v, (short)0, acc[tj], false, false);
}

// ---------------------------------------------------------------------------
// Gate GEMM, split-K x4: actp[ks](128x4096) = hy(128x3072 bf16) @ Wp slice.
// grid = 256 blocks (4 k-splits x 8 m-tiles x 8 j-groups), 256 thr = 8 waves.
// The block's shared 16-row A strip (24 KB) is staged once into LDS with
// global_load_async_to_lds_b128 (ASYNCcnt), then each wave reads A fragments
// via ds_load_b128 while the per-wave B stream stays register double-buffered
// from L2 so global_load_b128 of step kt+1 overlap the 4 v_wmma of step kt.
// ---------------------------------------------------------------------------
__global__ void gates_gemm_kernel(const __bf16* __restrict__ hy,
                                  const __bf16* __restrict__ Wp,
                                  float* __restrict__ actp) {
    __shared__ __align__(16) __bf16 Alds[16 * KW];     // 24 KB

    int wave = threadIdx.x >> 5;
    int lane = threadIdx.x & 31;
    int ks   = blockIdx.x & 3;          // k-split
    int mt   = (blockIdx.x >> 2) & 7;   // m tile (16 rows)
    int jg   = blockIdx.x >> 5;         // group of 512 columns
    int m0   = mt * 16;
    int j0   = jg * 512 + wave * 64;    // this wave's 64 columns
    long jt0 = j0 >> 4;
    int half = (lane >= 16) ? 1 : 0;
    int ncol = lane & 15;
    const int k0 = ks * KTP;

    // ---- cooperative async stage of the A strip: rows m0..m0+15, K k0*32.. ----
    {
        unsigned lds_base = (unsigned)(unsigned long long)(size_t)&Alds[0];
        const char* gbase = (const char*)(hy + (long)m0 * KC + k0 * 32);
        // 16 rows * 768 bf16 = 1536 chunks of 16B; 6 chunks per thread
#pragma unroll
        for (int q = 0; q < 6; ++q) {
            int c   = threadIdx.x + q * 256;
            int row = c / 96;               // 96 x 16B chunks per row
            int cc  = c % 96;
            async_copy_b128(lds_base + (unsigned)(row * (KW * 2) + cc * 16),
                            gbase + (long)row * (KC * 2) + cc * 16);
        }
        wait_async0();
        __syncthreads();
    }
    const __bf16* alrow = Alds + (lane & 15) * KW;

    v8f acc[4];
#pragma unroll
    for (int tj = 0; tj < 4; ++tj)
#pragma unroll
        for (int r = 0; r < 8; ++r) acc[tj][r] = 0.0f;

    FragU a0, a1, b0[4], b1[4];
    load_a_lds(a0, alrow, 0, half);
    load_b4(b0, Wp, jt0, k0, lane);

#pragma unroll 1
    for (int it = 0; it < KTP; it += 2) {
        // prefetch (via L1) ahead on the weight stream
        __builtin_prefetch(Wp + ((jt0 * KT32 + (long)(k0 + ((it + 3) % KTP))) << 9) + lane * 16, 0, 1);
        // stage kt+1 while WMMAs for kt run
        load_a_lds(a1, alrow, it + 1, half);
        load_b4(b1, Wp, jt0, k0 + it + 1, lane);
        mm4(acc, a0, b0);
        if (it + 2 < KTP) {
            load_a_lds(a0, alrow, it + 2, half);
            load_b4(b0, Wp, jt0, k0 + it + 2, lane);
        }
        mm4(acc, a1, b1);
    }

    // C layout: VGPR r -> row m0 + r + 8*half, col j0 + tj*16 + (lane&15)
    float* actb = actp + (long)ks * NB * FH;
#pragma unroll
    for (int tj = 0; tj < 4; ++tj) {
        long col = j0 + tj * 16 + ncol;
#pragma unroll
        for (int r = 0; r < 8; ++r) {
            long row = m0 + r + (half ? 8 : 0);
            actb[row * FH + col] = acc[tj][r];
        }
    }
}

// ---------------------------------------------------------------------------
// Elementwise gates + state update (sums the 4 K-split partials + bias);
// writes h into out[n, t, :]
// ---------------------------------------------------------------------------
__global__ void gates_ew_kernel(const float* __restrict__ actp,
                                const float* __restrict__ bias,
                                float* __restrict__ cbuf,
                                float* __restrict__ hbuf,
                                float* __restrict__ out,
                                int t) {
    long idx = (long)blockIdx.x * 256 + threadIdx.x;   // n*H + j
    long n = idx >> 10;
    long j = idx & 1023;
    float g4[4];
#pragma unroll
    for (int q = 0; q < 4; ++q) {
        float s = bias[q * HH + j];
#pragma unroll
        for (int ks = 0; ks < KSPLIT; ++ks)
            s += actp[((long)ks * NB + n) * FH + q * HH + j];
        g4[q] = s;
    }
    float iv = sigmoidf(g4[0]);
    float fv = sigmoidf(g4[1]);
    float ov = sigmoidf(g4[2]);
    float gv = tanhf(g4[3]);
    float c  = fv * cbuf[idx] + iv * gv;
    cbuf[idx] = c;
    float h = ov * tanhf(c);
    hbuf[idx] = h;
    out[(n * TT + t) * HH + j] = h;
}

// ---------------------------------------------------------------------------
extern "C" void kernel_launch(void* const* d_in, const int* in_sizes, int n_in,
                              void* d_out, int out_size, void* d_ws, size_t ws_size,
                              hipStream_t stream) {
    const float* x     = (const float*)d_in[0];
    const float* Af    = (const float*)d_in[1];   // (N,H,4,4) == (N,H,16)
    const float* Wx    = (const float*)d_in[2];
    const float* Wh    = (const float*)d_in[3];
    const float* Wattn = (const float*)d_in[4];
    const float* bias  = (const float*)d_in[5];
    float* out = (float*)d_out;

    char* ws = (char*)d_ws;
    size_t off = 0;
    __bf16* Wp  = (__bf16*)(ws + off); off += (size_t)KC * FH * 2;            // 25.2 MB
    float* hbuf = (float*)(ws + off);  off += (size_t)NB * HH * 4;            // 0.5 MB
    float* cbuf = (float*)(ws + off);  off += (size_t)NB * HH * 4;            // 0.5 MB
    __bf16* hy  = (__bf16*)(ws + off); off += (size_t)NB * KC * 2;            // 0.75 MB
    float* actp = (float*)(ws + off);  off += (size_t)KSPLIT * NB * FH * 4;   // 8 MB

    // One-time (per call) weight pack + state init
    pack_weights_kernel<<<(KC * FH) / 256, 256, 0, stream>>>(Wx, Wh, Wattn, Wp);
    init_state_kernel<<<(NB * HH) / 256, 256, 0, stream>>>(Af, hbuf, cbuf);

    for (int t = 0; t < TT; ++t) {
        attn_pack_kernel<<<NB, 256, 0, stream>>>(Af, x, hbuf, hy, t);
        gates_gemm_kernel<<<256, 256, 0, stream>>>(hy, Wp, actp);
        gates_ew_kernel<<<(NB * HH) / 256, 256, 0, stream>>>(actp, bias, cbuf, hbuf, out, t);
    }
}
